// EncoderLayer_41257455845724
// MI455X (gfx1250) — matrix operand order repaired and checked by reading, not compiled, not run on hardware
//
#include <hip/hip_runtime.h>
#include <hip/hip_bf16.h>
#include <math.h>

typedef __bf16 bf16_t;
typedef bf16_t v16bf __attribute__((ext_vector_type(16)));
typedef float  v8f   __attribute__((ext_vector_type(8)));
typedef unsigned int u32x4 __attribute__((ext_vector_type(4)));
typedef unsigned int u32x2 __attribute__((ext_vector_type(2)));

#define CB   32
#define CL   4096
#define CC   512
#define CWS  64
#define CSH  32
#define CNH  8
#define CDK  64
#define CHID 2048
#define CNW  (CL / CWS)          // 64
#define TOK  (CB * CL)           // 131072

// ---------------- helpers ----------------
__device__ __forceinline__ float gelu_exact(float x) {
    return 0.5f * x * (1.0f + erff(x * 0.70710678118654752440f));
}

__device__ __forceinline__ v8f vzero8() {
    v8f z = {0.f, 0.f, 0.f, 0.f, 0.f, 0.f, 0.f, 0.f};
    return z;
}

// A fragment: 16x32 bf16, rows m = lane&15, chunks at K = (lane>=16?8:0) and +16
__device__ __forceinline__ v16bf load_a16x32(const bf16_t* p0, int ld) {
    int lane = threadIdx.x & 31;
    const bf16_t* p = p0 + (size_t)(lane & 15) * ld + ((lane >> 4) << 3);
    v16bf a;
    u32x4* d = reinterpret_cast<u32x4*>(&a);
    d[0] = *reinterpret_cast<const u32x4*>(p);
    d[1] = *reinterpret_cast<const u32x4*>(p + 16);
    return a;
}

// B fragment: 32x16 bf16 from [N][K] row-major source: lane n = lane&15,
// chunks at K = (lane>=16?8:0) and +16
__device__ __forceinline__ v16bf load_b32x16(const bf16_t* p0, int ld) {
    int lane = threadIdx.x & 31;
    const bf16_t* p = p0 + (size_t)(lane & 15) * ld + ((lane >> 4) << 3);
    v16bf b;
    u32x4* d = reinterpret_cast<u32x4*>(&b);
    d[0] = *reinterpret_cast<const u32x4*>(p);
    d[1] = *reinterpret_cast<const u32x4*>(p + 16);
    return b;
}

__device__ __forceinline__ v8f wmma_bf16(v16bf a, v16bf b, v8f c) {
    return __builtin_amdgcn_wmma_f32_16x16x32_bf16(false, a, false, b, (short)0, c,
                                                   false, false);
}

// ---------------- f32 -> bf16 weight conversion ----------------
__global__ __launch_bounds__(256) void cvt_bf16_kernel(const float* __restrict__ s,
                                                       bf16_t* __restrict__ d, int n) {
    int i = blockIdx.x * 256 + threadIdx.x;
    if (i < n) d[i] = (bf16_t)s[i];
}

// ---------------- LayerNorm (+ optional roll) -> bf16 ----------------
// one wave (32 lanes) per token, 8 tokens per block
__global__ __launch_bounds__(256) void ln_kernel(const float* __restrict__ src,
                                                 const float* __restrict__ g,
                                                 const float* __restrict__ bb,
                                                 bf16_t* __restrict__ out, int shift) {
    int lane = threadIdx.x & 31;
    int wave = threadIdx.x >> 5;
    long gid = (long)blockIdx.x * 8 + wave;           // destination token
    long b = gid >> 12;
    int  t = (int)(gid & 4095);
    long srow = (b << 12) + ((t + shift) & 4095);     // rolled source token
    const float* p = src + srow * CC;

    float4 vals[4];
    float s = 0.f, sq = 0.f;
#pragma unroll
    for (int i = 0; i < 4; ++i) {
        vals[i] = reinterpret_cast<const float4*>(p)[lane + (i << 5)];
        s  += vals[i].x + vals[i].y + vals[i].z + vals[i].w;
        sq += vals[i].x * vals[i].x + vals[i].y * vals[i].y +
              vals[i].z * vals[i].z + vals[i].w * vals[i].w;
    }
#pragma unroll
    for (int o = 16; o > 0; o >>= 1) {
        s  += __shfl_xor(s,  o, 32);
        sq += __shfl_xor(sq, o, 32);
    }
    float mean = s * (1.0f / CC);
    float var  = sq * (1.0f / CC) - mean * mean;
    float rs   = rsqrtf(var + 1e-5f);

    bf16_t* q = out + gid * CC;
#pragma unroll
    for (int i = 0; i < 4; ++i) {
        int idx = lane + (i << 5);                    // float4 index
        float4 gg = reinterpret_cast<const float4*>(g)[idx];
        float4 bv = reinterpret_cast<const float4*>(bb)[idx];
        float4 v  = vals[i];
        __align__(8) bf16_t tmp[4];
        tmp[0] = (bf16_t)((v.x - mean) * rs * gg.x + bv.x);
        tmp[1] = (bf16_t)((v.y - mean) * rs * gg.y + bv.y);
        tmp[2] = (bf16_t)((v.z - mean) * rs * gg.z + bv.z);
        tmp[3] = (bf16_t)((v.w - mean) * rs * gg.w + bv.w);
        *reinterpret_cast<u32x2*>(&q[idx << 2]) = *reinterpret_cast<const u32x2*>(tmp);
    }
}

// ---------------- 128x128 tiled GEMM: out = A @ W^T (+epilogue) ----------------
// A: [M][K] bf16, W: [N][K] bf16 (torch Linear weight layout)
// MODE 0: out_h = result (bf16)                         (QKV projections)
// MODE 1: out_f[rolled] = result + resid[rolled] (f32)  (O-proj + roll + residual)
// MODE 2: out_h = gelu(result + bias) (bf16)            (FFN c1)
// MODE 3: out_f = result + bias + resid (f32)           (FFN c3 + residual -> d_out)
template <int MODE>
__global__ __launch_bounds__(256) void gemm128(const bf16_t* __restrict__ A,
                                               const bf16_t* __restrict__ W,
                                               int K, int N,
                                               const float* __restrict__ bias,
                                               const float* __restrict__ resid,
                                               float* __restrict__ out_f,
                                               bf16_t* __restrict__ out_h) {
    constexpr int LDS = 40;  // padded stride (bf16): conflict-free 16B frag reads
    __shared__ __align__(16) bf16_t sA[2][128 * LDS];
    __shared__ __align__(16) bf16_t sB[2][128 * LDS];

    const int tid  = threadIdx.x;
    const int lane = tid & 31;
    const int wave = tid >> 5;
    const int wm = (wave >> 2) * 64;   // 0 / 64
    const int wn = (wave & 3) * 32;    // 0 / 32 / 64 / 96
    const long rowBase = (long)blockIdx.x * 128;
    const int  colBase = blockIdx.y * 128;

    const int sr = tid >> 2;           // staging row (0..63, +64 on 2nd chunk)
    const int sc = (tid & 3) << 3;     // staging col within 32-wide slab

    v8f acc[4][2];
#pragma unroll
    for (int i = 0; i < 4; ++i) {
        acc[i][0] = vzero8();
        acc[i][1] = vzero8();
    }

    // prologue: stage k-slab 0
#pragma unroll
    for (int i = 0; i < 2; ++i) {
        int r = sr + i * 64;
        *reinterpret_cast<u32x4*>(&sA[0][r * LDS + sc]) =
            *reinterpret_cast<const u32x4*>(&A[(rowBase + r) * (long)K + sc]);
        *reinterpret_cast<u32x4*>(&sB[0][r * LDS + sc]) =
            *reinterpret_cast<const u32x4*>(&W[(long)(colBase + r) * K + sc]);
    }
    __syncthreads();

    int cur = 0;
    for (int k0 = 0; k0 < K; k0 += 32) {
        const bool hasNext = (k0 + 32 < K);
        u32x4 ra[2], rb[2];
        if (hasNext) {  // fetch next slab into registers (overlaps with WMMA below)
#pragma unroll
            for (int i = 0; i < 2; ++i) {
                int r = sr + i * 64;
                ra[i] = *reinterpret_cast<const u32x4*>(
                    &A[(rowBase + r) * (long)K + k0 + 32 + sc]);
                rb[i] = *reinterpret_cast<const u32x4*>(
                    &W[(long)(colBase + r) * K + k0 + 32 + sc]);
            }
            if (k0 + 64 < K) {  // prefetch two slabs ahead (global_prefetch_b8)
                __builtin_prefetch(&A[(rowBase + sr) * (long)K + k0 + 64], 0, 1);
                __builtin_prefetch(&W[(long)(colBase + sr) * K + k0 + 64], 0, 1);
            }
        }

        v16bf bfr[2];
#pragma unroll
        for (int nt = 0; nt < 2; ++nt)
            bfr[nt] = load_b32x16(&sB[cur][(wn + nt * 16) * LDS], LDS);
#pragma unroll
        for (int mt = 0; mt < 4; ++mt) {
            v16bf af = load_a16x32(&sA[cur][(wm + mt * 16) * LDS], LDS);
            acc[mt][0] = wmma_bf16(af, bfr[0], acc[mt][0]);
            acc[mt][1] = wmma_bf16(af, bfr[1], acc[mt][1]);
        }

        if (hasNext) {
            int nxt = cur ^ 1;
#pragma unroll
            for (int i = 0; i < 2; ++i) {
                int r = sr + i * 64;
                *reinterpret_cast<u32x4*>(&sA[nxt][r * LDS + sc]) = ra[i];
                *reinterpret_cast<u32x4*>(&sB[nxt][r * LDS + sc]) = rb[i];
            }
            __syncthreads();
            cur = nxt;
        }
    }

    // epilogue: D tile layout -> lane col = lane&15, vgpr r -> row r + (lane>=16)*8
    const int nn = lane & 15;
    const int mo = (lane >> 4) * 8;
#pragma unroll
    for (int mt = 0; mt < 4; ++mt) {
#pragma unroll
        for (int nt = 0; nt < 2; ++nt) {
            int n = colBase + wn + nt * 16 + nn;
#pragma unroll
            for (int r = 0; r < 8; ++r) {
                long m = rowBase + wm + mt * 16 + mo + r;
                float v = acc[mt][nt][r];
                if (MODE == 0) {
                    out_h[m * (long)N + n] = (bf16_t)v;
                } else if (MODE == 1) {
                    long b = m >> 12;
                    int  t = (int)(m & 4095);
                    long mo2 = (b << 12) + ((t + CSH) & 4095);  // roll back by +SHIFT
                    out_f[mo2 * CC + n] = v + resid[mo2 * CC + n];
                } else if (MODE == 2) {
                    out_h[m * (long)N + n] = (bf16_t)gelu_exact(v + bias[n]);
                } else {
                    out_f[m * CC + n] = v + bias[n] + resid[m * CC + n];
                }
            }
        }
    }
}

// ---------------- windowed attention: one block per (window, head) ----------------
__global__ __launch_bounds__(128) void attn_kernel(const bf16_t* __restrict__ Q,
                                                   const bf16_t* __restrict__ Kb,
                                                   const bf16_t* __restrict__ V,
                                                   const float* __restrict__ rel,
                                                   float* __restrict__ attn_out,
                                                   bf16_t* __restrict__ ctx) {
    __shared__ __align__(16) float  sS[64 * 65];
    __shared__ __align__(16) bf16_t sP[64 * 72];
    __shared__ __align__(16) bf16_t sVt[64 * 72];
    __shared__ float sBias[128];      // rel-pos bias for this head: index m-n+63

    const int tid = threadIdx.x, lane = tid & 31, wave = tid >> 5;
    const int head = blockIdx.x & 7;
    const long win = blockIdx.x >> 3;     // [0, B*nW)
    const int wi   = (int)(win & 63);     // window index within image
    const long tok0 = win * CWS;
    const int  col0 = head * CDK;

    // stage rel-pos bias (127 values for this head)
    if (tid < 127) sBias[tid] = rel[tid * CNH + head];

    // stage V^T (d-major) into LDS
    for (int c = tid; c < 512; c += 128) {
        int m  = c >> 3;
        int d0 = (c & 7) << 3;
        u32x4 vv = *reinterpret_cast<const u32x4*>(&V[(tok0 + m) * CC + col0 + d0]);
        const bf16_t* e = reinterpret_cast<const bf16_t*>(&vv);
#pragma unroll
        for (int j = 0; j < 8; ++j) sVt[(d0 + j) * 72 + m] = e[j];
    }
    __syncthreads();   // sBias (and sVt) visible

    // scores S = Q K^T : wave owns rows [wave*16, wave*16+16), all 64 cols
    const int m0 = wave * 16;
    v8f sacc[4];
#pragma unroll
    for (int i = 0; i < 4; ++i) sacc[i] = vzero8();
#pragma unroll
    for (int ks = 0; ks < 2; ++ks) {
        v16bf aq = load_a16x32(&Q[(tok0 + m0) * CC + col0 + ks * 32], CC);
#pragma unroll
        for (int nt = 0; nt < 4; ++nt) {
            v16bf bk = load_b32x16(&Kb[(tok0 + nt * 16) * CC + col0 + ks * 32], CC);
            sacc[nt] = wmma_bf16(aq, bk, sacc[nt]);
        }
    }
    // scale + relative-position bias + shift mask -> LDS
    const int nn = lane & 15;
    const int mo = (lane >> 4) * 8;
#pragma unroll
    for (int nt = 0; nt < 4; ++nt) {
        int n = nt * 16 + nn;
#pragma unroll
        for (int r = 0; r < 8; ++r) {
            int m = m0 + mo + r;
            float v = sacc[nt][r] * 0.125f + sBias[m - n + 63];
            if (wi == CNW - 1 && ((m < CSH) != (n < CSH))) v -= 100.0f;
            sS[m * 65 + n] = v;
        }
    }
    __syncthreads();

    // softmax over rows; emit attn (f32, tuple output) + P (bf16)
    if (tid < 64) {
        float mx = -1e30f;
        for (int n = 0; n < 64; ++n) mx = fmaxf(mx, sS[tid * 65 + n]);
        float sum = 0.f;
        for (int n = 0; n < 64; ++n) {
            float e = expf(sS[tid * 65 + n] - mx);
            sS[tid * 65 + n] = e;
            sum += e;
        }
        float inv = 1.0f / sum;
        long abase = ((long)blockIdx.x * 64 + tid) * 64;   // [(win*NH+head), row, col]
        for (int n4 = 0; n4 < 16; ++n4) {
            float4 pv;
            pv.x = sS[tid * 65 + n4 * 4 + 0] * inv;
            pv.y = sS[tid * 65 + n4 * 4 + 1] * inv;
            pv.z = sS[tid * 65 + n4 * 4 + 2] * inv;
            pv.w = sS[tid * 65 + n4 * 4 + 3] * inv;
            sP[tid * 72 + n4 * 4 + 0] = (bf16_t)pv.x;
            sP[tid * 72 + n4 * 4 + 1] = (bf16_t)pv.y;
            sP[tid * 72 + n4 * 4 + 2] = (bf16_t)pv.z;
            sP[tid * 72 + n4 * 4 + 3] = (bf16_t)pv.w;
            reinterpret_cast<float4*>(attn_out + abase)[n4] = pv;
        }
    }
    __syncthreads();

    // ctx = P @ V
    v8f cacc[4];
#pragma unroll
    for (int i = 0; i < 4; ++i) cacc[i] = vzero8();
#pragma unroll
    for (int ks = 0; ks < 2; ++ks) {
        v16bf ap = load_a16x32(&sP[m0 * 72 + ks * 32], 72);
#pragma unroll
        for (int nt = 0; nt < 4; ++nt) {
            v16bf bv = load_b32x16(&sVt[(nt * 16) * 72 + ks * 32], 72);
            cacc[nt] = wmma_bf16(ap, bv, cacc[nt]);
        }
    }
#pragma unroll
    for (int nt = 0; nt < 4; ++nt) {
#pragma unroll
        for (int r = 0; r < 8; ++r) {
            int m = m0 + mo + r;
            ctx[(tok0 + m) * CC + col0 + nt * 16 + nn] = (bf16_t)cacc[nt][r];
        }
    }
}

// ---------------- depthwise width-3 conv over L + skip + GELU ----------------
__global__ __launch_bounds__(256) void conv_ffn_kernel(const bf16_t* __restrict__ f,
                                                       const float* __restrict__ w2,
                                                       const float* __restrict__ b2,
                                                       bf16_t* __restrict__ y) {
    long gid = blockIdx.x;              // token
    int  t   = (int)(gid & 4095);
    int  c0  = threadIdx.x << 3;
    const bf16_t* fr = f + gid * CHID + c0;

    u32x4 vc = *reinterpret_cast<const u32x4*>(fr);
    u32x4 vm = {0u, 0u, 0u, 0u}, vp = {0u, 0u, 0u, 0u};
    if (t > 0)    vm = *reinterpret_cast<const u32x4*>(fr - CHID);
    if (t < 4095) vp = *reinterpret_cast<const u32x4*>(fr + CHID);
    const bf16_t* ec = reinterpret_cast<const bf16_t*>(&vc);
    const bf16_t* em = reinterpret_cast<const bf16_t*>(&vm);
    const bf16_t* ep = reinterpret_cast<const bf16_t*>(&vp);

    __align__(16) bf16_t outv[8];
#pragma unroll
    for (int j = 0; j < 8; ++j) {
        int c = c0 + j;
        float fc = (float)ec[j];
        float dw = (float)em[j] * w2[c * 3 + 0] + fc * w2[c * 3 + 1] +
                   (float)ep[j] * w2[c * 3 + 2] + b2[c];
        outv[j] = (bf16_t)gelu_exact(dw + fc);
    }
    *reinterpret_cast<u32x4*>(&y[gid * CHID + c0]) =
        *reinterpret_cast<const u32x4*>(outv);
}

// ---------------- launch ----------------
extern "C" void kernel_launch(void* const* d_in, const int* in_sizes, int n_in,
                              void* d_out, int out_size, void* d_ws, size_t ws_size,
                              hipStream_t stream) {
    const float* x     = (const float*)d_in[0];
    const float* ln1_g = (const float*)d_in[1];
    const float* ln1_b = (const float*)d_in[2];
    const float* w_q   = (const float*)d_in[3];
    const float* w_k   = (const float*)d_in[4];
    const float* w_v   = (const float*)d_in[5];
    const float* w_o   = (const float*)d_in[6];
    const float* rel   = (const float*)d_in[7];
    const float* ln2_g = (const float*)d_in[8];
    const float* ln2_b = (const float*)d_in[9];
    const float* c1_w  = (const float*)d_in[10];
    const float* c1_b  = (const float*)d_in[11];
    const float* c2_w  = (const float*)d_in[12];
    const float* c2_b  = (const float*)d_in[13];
    const float* c3_w  = (const float*)d_in[14];
    const float* c3_b  = (const float*)d_in[15];

    char* ws = (char*)d_ws;
    size_t o = 0;
    auto take = [&](size_t bytes) { char* p = ws + o; o += (bytes + 255) & ~(size_t)255; return p; };

    bf16_t* wq_h  = (bf16_t*)take((size_t)CC * CC * 2);
    bf16_t* wk_h  = (bf16_t*)take((size_t)CC * CC * 2);
    bf16_t* wv_h  = (bf16_t*)take((size_t)CC * CC * 2);
    bf16_t* wo_h  = (bf16_t*)take((size_t)CC * CC * 2);
    bf16_t* c1w_h = (bf16_t*)take((size_t)CHID * CC * 2);
    bf16_t* c3w_h = (bf16_t*)take((size_t)CC * CHID * 2);
    bf16_t* hbuf  = (bf16_t*)take((size_t)TOK * CC * 2);      // LN1 out; reused for LN2 out
    bf16_t* qbuf  = (bf16_t*)take((size_t)TOK * CC * 2);
    bf16_t* kbuf  = (bf16_t*)take((size_t)TOK * CC * 2);
    bf16_t* vbuf  = (bf16_t*)take((size_t)TOK * CC * 2);
    bf16_t* ctxb  = (bf16_t*)take((size_t)TOK * CC * 2);
    float*  res1  = (float*)take((size_t)TOK * CC * 4);
    bf16_t* fbuf  = qbuf;                                      // alias q|k|v|ctx (512MB)
    bf16_t* ybuf  = (bf16_t*)take((size_t)TOK * CHID * 2);

    float* out_main = (float*)d_out;
    float* out_attn = (float*)d_out + (size_t)CB * CL * CC;

    // weight conversions
    cvt_bf16_kernel<<<(CC * CC + 255) / 256, 256, 0, stream>>>(w_q, wq_h, CC * CC);
    cvt_bf16_kernel<<<(CC * CC + 255) / 256, 256, 0, stream>>>(w_k, wk_h, CC * CC);
    cvt_bf16_kernel<<<(CC * CC + 255) / 256, 256, 0, stream>>>(w_v, wv_h, CC * CC);
    cvt_bf16_kernel<<<(CC * CC + 255) / 256, 256, 0, stream>>>(w_o, wo_h, CC * CC);
    cvt_bf16_kernel<<<(CHID * CC + 255) / 256, 256, 0, stream>>>(c1_w, c1w_h, CHID * CC);
    cvt_bf16_kernel<<<(CC * CHID + 255) / 256, 256, 0, stream>>>(c3_w, c3w_h, CC * CHID);

    // LN1 + roll(-SHIFT) -> bf16
    ln_kernel<<<TOK / 8, 256, 0, stream>>>(x, ln1_g, ln1_b, hbuf, CSH);

    // QKV projections
    dim3 g512(TOK / 128, CC / 128);
    gemm128<0><<<g512, 256, 0, stream>>>(hbuf, wq_h, CC, CC, nullptr, nullptr, nullptr, qbuf);
    gemm128<0><<<g512, 256, 0, stream>>>(hbuf, wk_h, CC, CC, nullptr, nullptr, nullptr, kbuf);
    gemm128<0><<<g512, 256, 0, stream>>>(hbuf, wv_h, CC, CC, nullptr, nullptr, nullptr, vbuf);

    // attention (also writes attn tuple output in f32)
    attn_kernel<<<CB * CNW * CNH, 128, 0, stream>>>(qbuf, kbuf, vbuf, rel, out_attn, ctxb);

    // O-projection + roll(+SHIFT) + residual -> res1 (f32)
    gemm128<1><<<g512, 256, 0, stream>>>(ctxb, wo_h, CC, CC, nullptr, x, res1, nullptr);

    // LN2 -> bf16 (reuse hbuf)
    ln_kernel<<<TOK / 8, 256, 0, stream>>>(res1, ln2_g, ln2_b, hbuf, 0);

    // FFN c1 + GELU -> fbuf (bf16, aliases q|k|v|ctx)
    dim3 g2048(TOK / 128, CHID / 128);
    gemm128<2><<<g2048, 256, 0, stream>>>(hbuf, c1w_h, CC, CHID, c1_b, nullptr, nullptr, fbuf);

    // depthwise conv + skip + GELU -> ybuf
    conv_ffn_kernel<<<TOK, 256, 0, stream>>>(fbuf, c2_w, c2_b, ybuf);

    // FFN c3 + bias + residual -> d_out (f32)
    gemm128<3><<<g512, 256, 0, stream>>>(ybuf, c3w_h, CHID, CC, c3_b, res1, out_main, nullptr);
}
